// NodeBlock_6751688589928
// MI455X (gfx1250) — matrix operand order repaired and checked by reading, not compiled, not run on hardware
//
#include <hip/hip_runtime.h>

// ---------- types ----------
typedef _Float16 v4h  __attribute__((ext_vector_type(4)));
typedef _Float16 v8h  __attribute__((ext_vector_type(8)));
typedef _Float16 v16h __attribute__((ext_vector_type(16)));
typedef float    v4f  __attribute__((ext_vector_type(4)));
typedef float    v8f  __attribute__((ext_vector_type(8)));

#define WAVES 4
#define TB    (WAVES * 32)
#define KD1   136   // 128 + 8 halves pad (row = 272 B, 16B aligned)
#define KD2   72    // 64 + 8 halves pad  (row = 144 B, 16B aligned)
#define EPS   1e-5f

// Build a v16h A/B fragment from two 16B LDS chunks.
__device__ __forceinline__ v16h frag16(const _Float16* p0, const _Float16* p1) {
  v8h lo = *(const v8h*)p0;
  v8h hi = *(const v8h*)p1;
  return __builtin_shufflevector(lo, hi, 0,1,2,3,4,5,6,7,8,9,10,11,12,13,14,15);
}

__device__ __forceinline__ v8f wmma16(v16h a, v16h b, v8f c) {
  return __builtin_amdgcn_wmma_f32_16x16x32_f16(false, a, false, b, (short)0, c, false, false);
}

__device__ __forceinline__ void stage4(_Float16* dst, const float* src, float s) {
  v4f f = *(const v4f*)src;
  *(v4h*)dst = (v4h){(_Float16)(f.x*s), (_Float16)(f.y*s), (_Float16)(f.z*s), (_Float16)(f.w*s)};
}
__device__ __forceinline__ void lds_fence() {
  asm volatile("s_wait_dscnt 0" ::: "memory");
}
// packed-f16 BN + ReLU on 8 lanes' worth of K values
__device__ __forceinline__ v8h bnrelu8(v8h h, v8h sc, v8h sh) {
  v8h y = h * sc + sh;
  v8h z = {};
  return __builtin_elementwise_max(y, z);
}

// =====================================================================
// Edge layer-1 kernels over FULL 16-edge tiles.
//   MODE 0: BN batch stats only (recompute strategy fallback)
//   MODE 1: BN batch stats + store h1 as f16 [E][64]   (cached strategy)
//   MODE 2: full recompute apply: layer1+BN+ReLU+layer2+scatter (fallback)
// =====================================================================
template <int MODE>
__global__ __launch_bounds__(TB) void k_edge(
    const float* __restrict__ x, const int* __restrict__ rowp,
    const int* __restrict__ colp, const float* __restrict__ ea,
    const float* __restrict__ W1, const float* __restrict__ b1,
    const float* __restrict__ W2, const float* __restrict__ b2,
    const float* __restrict__ scale, const float* __restrict__ shift,
    float* __restrict__ gsum, float* __restrict__ gssq,
    float* __restrict__ summed, int* __restrict__ cnt,
    _Float16* __restrict__ h1buf, int E) {
  constexpr bool STATS = (MODE != 2);
  __shared__ _Float16 Wt1[64][KD1];        // W1 transposed: Wt1[n][k]
  __shared__ _Float16 Wt2[64][KD2];        // W2 transposed (MODE 2)
  __shared__ _Float16 As[WAVES][16][KD1];  // per-wave input tile (f16)
  __shared__ _Float16 Ys[WAVES][16][KD2];  // per-wave h1/ReLU tile (MODE 1/2)
  __shared__ float lsum[64], lssq[64];

  const int tid  = threadIdx.x;
  const int lane = tid & 31;
  const int wave = __builtin_amdgcn_readfirstlane(tid >> 5);  // SGPR
  const int m = lane & 15, hi = lane >> 4, kh = hi << 3;

  for (int i = tid; i < 128 * 64; i += TB) Wt1[i & 63][i >> 6] = (_Float16)W1[i];
  if (MODE == 2)
    for (int i = tid; i < 64 * 64; i += TB) Wt2[i & 63][i >> 6] = (_Float16)W2[i];
  if (STATS && tid < 64) { lsum[tid] = 0.f; lssq[tid] = 0.f; }
  __syncthreads();

  float bj[4], scj[4], shj[4], b2j[4];
  #pragma unroll
  for (int j = 0; j < 4; ++j) {
    const int n = (j << 4) + m;
    bj[j] = b1[n];
    if (MODE == 2) { scj[j] = scale[n]; shj[j] = shift[n]; b2j[j] = b2[n]; }
  }
  const _Float16* Ar = &As[wave][m][0];
  const _Float16* Yr = &Ys[wave][m][0];
  const _Float16* Br1[4];
  const _Float16* Br2[4];
  #pragma unroll
  for (int j = 0; j < 4; ++j) {
    Br1[j] = &Wt1[(j << 4) + m][hi << 3];
    if (MODE == 2) Br2[j] = &Wt2[(j << 4) + m][hi << 3];
  }

  float accS[4] = {0.f, 0.f, 0.f, 0.f};
  float accQ[4] = {0.f, 0.f, 0.f, 0.f};

  const int ntiles = E >> 4;                 // full tiles only
  const int nw  = gridDim.x * WAVES;
  const int wid = blockIdx.x * WAVES + wave; // scalar

  for (int t = wid; t < ntiles; t += nw) {
    const int ebase = t << 4;
    // -------- stage concat(x[row[e]], ea[e]) -> f16 LDS tile --------
    #pragma unroll
    for (int i = 0; i < 16; ++i) {
      const int e = ebase + i;
      const float* src = (lane < 16)
          ? (x + (size_t)rowp[e] * 64 + (lane << 2))
          : (ea + (size_t)e * 64 + ((lane & 15) << 2));
      stage4(&As[wave][i][lane << 2], src, 1.f);
    }
    if (t + nw < ntiles)
      __builtin_prefetch(ea + ((size_t)(t + nw) << 4) * 64, 0, 1);
    lds_fence();

    // -------- layer 1: 4 K-chunks x 4 N-tiles of WMMA --------
    v16h a0 = frag16(Ar + 0  + kh, Ar + 16  + kh);
    v16h a1 = frag16(Ar + 32 + kh, Ar + 48  + kh);
    v16h a2 = frag16(Ar + 64 + kh, Ar + 80  + kh);
    v16h a3 = frag16(Ar + 96 + kh, Ar + 112 + kh);

    #pragma unroll
    for (int j = 0; j < 4; ++j) {
      const _Float16* Br = Br1[j];
      v8f acc;
      #pragma unroll
      for (int v = 0; v < 8; ++v) acc[v] = bj[j];
      acc = wmma16(a0, frag16(Br + 0,  Br + 16),  acc);
      acc = wmma16(a1, frag16(Br + 32, Br + 48),  acc);
      acc = wmma16(a2, frag16(Br + 64, Br + 80),  acc);
      acc = wmma16(a3, frag16(Br + 96, Br + 112), acc);

      const int n = (j << 4) + m;
      if (STATS) {
        #pragma unroll
        for (int v = 0; v < 8; ++v) {
          accS[j] += acc[v];
          accQ[j] = fmaf(acc[v], acc[v], accQ[j]);
        }
        if (MODE == 1) {
          #pragma unroll
          for (int v = 0; v < 8; ++v)
            Ys[wave][v + kh][n] = (_Float16)acc[v];      // raw h1
        }
      } else {  // MODE 2
        #pragma unroll
        for (int v = 0; v < 8; ++v) {
          float y = fmaxf(acc[v] * scj[j] + shj[j], 0.f);
          Ys[wave][v + kh][n] = (_Float16)y;
        }
      }
    }

    if (MODE == 1) {
      lds_fence();
      // coalesced f16 h1 store: 128 chunks of 8 halves, 4 per lane
      #pragma unroll
      for (int c = 0; c < 4; ++c) {
        const int chunk = lane + (c << 5);
        const int row = chunk >> 3, col8 = (chunk & 7) << 3;
        *(v8h*)(h1buf + ((size_t)(ebase + row) << 6) + col8) =
            *(const v8h*)(&Ys[wave][row][col8]);
      }
    }

    if (MODE == 2) {
      if (lane < 16) atomicAdd(&cnt[colp[ebase + lane]], 1);
      lds_fence();
      v16h y0 = frag16(Yr + 0  + kh, Yr + 16 + kh);
      v16h y1 = frag16(Yr + 32 + kh, Yr + 48 + kh);
      int dst[8];
      #pragma unroll
      for (int v = 0; v < 8; ++v) dst[v] = colp[ebase + v + kh];
      #pragma unroll
      for (int j = 0; j < 4; ++j) {
        const int n = (j << 4) + m;
        const _Float16* Br = Br2[j];
        v8f acc;
        #pragma unroll
        for (int v = 0; v < 8; ++v) acc[v] = b2j[j];
        acc = wmma16(y0, frag16(Br + 0,  Br + 16), acc);
        acc = wmma16(y1, frag16(Br + 32, Br + 48), acc);
        #pragma unroll
        for (int v = 0; v < 8; ++v)
          unsafeAtomicAdd(summed + (size_t)dst[v] * 64 + n, acc[v]);
      }
    }
  }

  if (STATS) {
    #pragma unroll
    for (int j = 0; j < 4; ++j) {
      const int n = (j << 4) + m;
      unsafeAtomicAdd(&lsum[n], accS[j]);
      unsafeAtomicAdd(&lssq[n], accQ[j]);
    }
    __syncthreads();
    if (tid < 64) {
      unsafeAtomicAdd(&gsum[tid], lsum[tid]);
      unsafeAtomicAdd(&gssq[tid], lssq[tid]);
    }
  }
}

// =====================================================================
// Cached edge pass 2: read f16 h1 tiles straight from global in A-frag
// layout, packed-f16 BN+ReLU (no LDS staging, no layer-1 recompute),
// layer-2 WMMA, scatter-add.
// =====================================================================
__global__ __launch_bounds__(TB) void k_edge2(
    const _Float16* __restrict__ h1buf, const int* __restrict__ colp,
    const float* __restrict__ W2, const float* __restrict__ b2,
    const float* __restrict__ scale, const float* __restrict__ shift,
    float* __restrict__ summed, int* __restrict__ cnt, int E) {
  __shared__ _Float16 Wt2[64][KD2];

  const int tid  = threadIdx.x;
  const int lane = tid & 31;
  const int wave = __builtin_amdgcn_readfirstlane(tid >> 5);
  const int m = lane & 15, hi = lane >> 4, kh = hi << 3;

  for (int i = tid; i < 64 * 64; i += TB) Wt2[i & 63][i >> 6] = (_Float16)W2[i];
  __syncthreads();

  // per-lane packed BN constants: piece p covers K = p*16 + kh + (0..7)
  v8h scv[4], shv[4];
  #pragma unroll
  for (int p = 0; p < 4; ++p)
    #pragma unroll
    for (int q = 0; q < 8; ++q) {
      const int k = (p << 4) + kh + q;
      scv[p][q] = (_Float16)scale[k];
      shv[p][q] = (_Float16)shift[k];
    }
  float b2j[4];
  const _Float16* Br2[4];
  #pragma unroll
  for (int j = 0; j < 4; ++j) {
    b2j[j] = b2[(j << 4) + m];
    Br2[j] = &Wt2[(j << 4) + m][hi << 3];
  }

  const int ntiles = E >> 4;
  const int nw  = gridDim.x * WAVES;
  const int wid = blockIdx.x * WAVES + wave;

  for (int t = wid; t < ntiles; t += nw) {
    const int ebase = t << 4;
    const _Float16* hr = h1buf + ((size_t)(ebase + m) << 6);
    v8h p0 = *(const v8h*)(hr + 0  + kh);
    v8h p1 = *(const v8h*)(hr + 16 + kh);
    v8h p2 = *(const v8h*)(hr + 32 + kh);
    v8h p3 = *(const v8h*)(hr + 48 + kh);
    if (t + nw < ntiles)
      __builtin_prefetch(h1buf + ((size_t)(t + nw) << 10), 0, 1);

    p0 = bnrelu8(p0, scv[0], shv[0]);
    p1 = bnrelu8(p1, scv[1], shv[1]);
    p2 = bnrelu8(p2, scv[2], shv[2]);
    p3 = bnrelu8(p3, scv[3], shv[3]);
    v16h y0 = __builtin_shufflevector(p0, p1, 0,1,2,3,4,5,6,7,8,9,10,11,12,13,14,15);
    v16h y1 = __builtin_shufflevector(p2, p3, 0,1,2,3,4,5,6,7,8,9,10,11,12,13,14,15);

    if (lane < 16) atomicAdd(&cnt[colp[ebase + lane]], 1);
    int dst[8];
    #pragma unroll
    for (int v = 0; v < 8; ++v) dst[v] = colp[ebase + v + kh];

    #pragma unroll
    for (int j = 0; j < 4; ++j) {
      const int n = (j << 4) + m;
      const _Float16* Br = Br2[j];
      v8f acc;
      #pragma unroll
      for (int v = 0; v < 8; ++v) acc[v] = b2j[j];
      acc = wmma16(y0, frag16(Br + 0,  Br + 16), acc);
      acc = wmma16(y1, frag16(Br + 32, Br + 48), acc);
      #pragma unroll
      for (int v = 0; v < 8; ++v)
        unsafeAtomicAdd(summed + (size_t)dst[v] * 64 + n, acc[v]);
    }
  }
}

// =====================================================================
// Node hot kernels: FULL 16-node tiles only.
// =====================================================================
template <bool APPLY>
__global__ __launch_bounds__(TB) void k_node(
    const float* __restrict__ x, const float* __restrict__ summed,
    const int* __restrict__ cnt,
    const float* __restrict__ W1, const float* __restrict__ b1,
    const float* __restrict__ W2, const float* __restrict__ b2,
    const float* __restrict__ scale, const float* __restrict__ shift,
    float* __restrict__ gsum, float* __restrict__ gssq,
    float* __restrict__ out, int N) {
  __shared__ _Float16 Wt1[64][KD1];
  __shared__ _Float16 Wt2[64][KD2];
  __shared__ _Float16 As[WAVES][16][KD1];  // reused as f32 [16][68] out-stage
  __shared__ _Float16 Ys[WAVES][16][KD2];
  __shared__ float lsum[64], lssq[64];

  const int tid  = threadIdx.x;
  const int lane = tid & 31;
  const int wave = __builtin_amdgcn_readfirstlane(tid >> 5);
  const int m = lane & 15, hi = lane >> 4, kh = hi << 3;

  for (int i = tid; i < 128 * 64; i += TB) Wt1[i & 63][i >> 6] = (_Float16)W1[i];
  if (APPLY)
    for (int i = tid; i < 64 * 64; i += TB) Wt2[i & 63][i >> 6] = (_Float16)W2[i];
  if (!APPLY && tid < 64) { lsum[tid] = 0.f; lssq[tid] = 0.f; }
  __syncthreads();

  float bj[4], scj[4], shj[4], b2j[4];
  #pragma unroll
  for (int j = 0; j < 4; ++j) {
    const int n = (j << 4) + m;
    bj[j] = b1[n];
    if (APPLY) { scj[j] = scale[n]; shj[j] = shift[n]; b2j[j] = b2[n]; }
  }
  const _Float16* Ar = &As[wave][m][0];
  const _Float16* Yr = &Ys[wave][m][0];
  const _Float16* Br1[4];
  const _Float16* Br2[4];
  #pragma unroll
  for (int j = 0; j < 4; ++j) {
    Br1[j] = &Wt1[(j << 4) + m][hi << 3];
    if (APPLY) Br2[j] = &Wt2[(j << 4) + m][hi << 3];
  }

  float accS[4] = {0.f, 0.f, 0.f, 0.f};
  float accQ[4] = {0.f, 0.f, 0.f, 0.f};

  const int ntiles = N >> 4;
  const int nw  = gridDim.x * WAVES;
  const int wid = blockIdx.x * WAVES + wave;

  for (int t = wid; t < ntiles; t += nw) {
    const int nbase = t << 4;
    #pragma unroll
    for (int i = 0; i < 16; ++i) {
      const int node = nbase + i;
      const float inv = (lane < 16) ? 1.f : (1.f / fmaxf((float)cnt[node], 1.f));
      const float* src = (lane < 16)
          ? (x + (size_t)node * 64 + (lane << 2))
          : (summed + (size_t)node * 64 + ((lane & 15) << 2));
      stage4(&As[wave][i][lane << 2], src, inv);
    }
    lds_fence();

    v16h a0 = frag16(Ar + 0  + kh, Ar + 16  + kh);
    v16h a1 = frag16(Ar + 32 + kh, Ar + 48  + kh);
    v16h a2 = frag16(Ar + 64 + kh, Ar + 80  + kh);
    v16h a3 = frag16(Ar + 96 + kh, Ar + 112 + kh);

    #pragma unroll
    for (int j = 0; j < 4; ++j) {
      const _Float16* Br = Br1[j];
      v8f acc;
      #pragma unroll
      for (int v = 0; v < 8; ++v) acc[v] = bj[j];
      acc = wmma16(a0, frag16(Br + 0,  Br + 16),  acc);
      acc = wmma16(a1, frag16(Br + 32, Br + 48),  acc);
      acc = wmma16(a2, frag16(Br + 64, Br + 80),  acc);
      acc = wmma16(a3, frag16(Br + 96, Br + 112), acc);

      if (!APPLY) {
        #pragma unroll
        for (int v = 0; v < 8; ++v) {
          accS[j] += acc[v];
          accQ[j] = fmaf(acc[v], acc[v], accQ[j]);
        }
      } else {
        const int n = (j << 4) + m;
        #pragma unroll
        for (int v = 0; v < 8; ++v) {
          float y = fmaxf(acc[v] * scj[j] + shj[j], 0.f);
          Ys[wave][v + kh][n] = (_Float16)y;
        }
      }
    }

    if (APPLY) {
      lds_fence();
      v16h y0 = frag16(Yr + 0  + kh, Yr + 16 + kh);
      v16h y1 = frag16(Yr + 32 + kh, Yr + 48 + kh);
      float* Fs = (float*)&As[wave][0][0];   // As tile dead: f32 [16][68] stage
      #pragma unroll
      for (int j = 0; j < 4; ++j) {
        const int n = (j << 4) + m;
        const _Float16* Br = Br2[j];
        v8f acc;
        #pragma unroll
        for (int v = 0; v < 8; ++v) acc[v] = b2j[j];
        acc = wmma16(y0, frag16(Br + 0,  Br + 16), acc);
        acc = wmma16(y1, frag16(Br + 32, Br + 48), acc);
        #pragma unroll
        for (int v = 0; v < 8; ++v)
          Fs[(v + kh) * 68 + n] = acc[v];
      }
      lds_fence();
      #pragma unroll
      for (int c = 0; c < 8; ++c) {
        const int chunk = lane + (c << 5);
        const int row = chunk >> 4, col4 = (chunk & 15) << 2;
        *(v4f*)(out + (size_t)(nbase + row) * 64 + col4) =
            *(const v4f*)(Fs + row * 68 + col4);
      }
    }
  }

  if (!APPLY) {
    #pragma unroll
    for (int j = 0; j < 4; ++j) {
      const int n = (j << 4) + m;
      unsafeAtomicAdd(&lsum[n], accS[j]);
      unsafeAtomicAdd(&lssq[n], accQ[j]);
    }
    __syncthreads();
    if (tid < 64) {
      unsafeAtomicAdd(&gsum[tid], lsum[tid]);
      unsafeAtomicAdd(&gssq[tid], lssq[tid]);
    }
  }
}

// =====================================================================
// Scalar tail kernels (<=15 rows); not launched when sizes divide by 16.
// =====================================================================
template <bool APPLY>
__global__ __launch_bounds__(64) void k_edge_tail(
    const float* __restrict__ x, const int* __restrict__ rowp,
    const int* __restrict__ colp, const float* __restrict__ ea,
    const float* __restrict__ W1, const float* __restrict__ b1,
    const float* __restrict__ W2, const float* __restrict__ b2,
    const float* __restrict__ scale, const float* __restrict__ shift,
    float* __restrict__ gsum, float* __restrict__ gssq,
    float* __restrict__ summed, int* __restrict__ cnt, int e0, int E) {
  __shared__ float in[128], y[64];
  const int n = threadIdx.x;
  float sl = 0.f, sq = 0.f;
  for (int e = e0; e < E; ++e) {
    in[n]      = x[(size_t)rowp[e] * 64 + n];
    in[64 + n] = ea[(size_t)e * 64 + n];
    __syncthreads();
    float h = b1[n];
    for (int k = 0; k < 128; ++k) h = fmaf(in[k], W1[k * 64 + n], h);
    if (!APPLY) {
      sl += h; sq = fmaf(h, h, sq);
    } else {
      y[n] = fmaxf(h * scale[n] + shift[n], 0.f);
      __syncthreads();
      float h2 = b2[n];
      for (int k = 0; k < 64; ++k) h2 = fmaf(y[k], W2[k * 64 + n], h2);
      unsafeAtomicAdd(summed + (size_t)colp[e] * 64 + n, h2);
      if (n == 0) atomicAdd(&cnt[colp[e]], 1);
    }
    __syncthreads();
  }
  if (!APPLY) { unsafeAtomicAdd(&gsum[n], sl); unsafeAtomicAdd(&gssq[n], sq); }
}

template <bool APPLY>
__global__ __launch_bounds__(64) void k_node_tail(
    const float* __restrict__ x, const float* __restrict__ summed,
    const int* __restrict__ cnt,
    const float* __restrict__ W1, const float* __restrict__ b1,
    const float* __restrict__ W2, const float* __restrict__ b2,
    const float* __restrict__ scale, const float* __restrict__ shift,
    float* __restrict__ gsum, float* __restrict__ gssq,
    float* __restrict__ out, int n0, int N) {
  __shared__ float in[128], y[64];
  const int n = threadIdx.x;
  float sl = 0.f, sq = 0.f;
  for (int node = n0; node < N; ++node) {
    const float inv = 1.f / fmaxf((float)cnt[node], 1.f);
    in[n]      = x[(size_t)node * 64 + n];
    in[64 + n] = summed[(size_t)node * 64 + n] * inv;
    __syncthreads();
    float h = b1[n];
    for (int k = 0; k < 128; ++k) h = fmaf(in[k], W1[k * 64 + n], h);
    if (!APPLY) {
      sl += h; sq = fmaf(h, h, sq);
    } else {
      y[n] = fmaxf(h * scale[n] + shift[n], 0.f);
      __syncthreads();
      float h2 = b2[n];
      for (int k = 0; k < 64; ++k) h2 = fmaf(y[k], W2[k * 64 + n], h2);
      out[(size_t)node * 64 + n] = h2;
    }
    __syncthreads();
  }
  if (!APPLY) { unsafeAtomicAdd(&gsum[n], sl); unsafeAtomicAdd(&gssq[n], sq); }
}

// ---------- BN stat finalize ----------
__global__ void k_finalize(const float* __restrict__ gsum, const float* __restrict__ gssq,
                           const float* __restrict__ g, const float* __restrict__ be,
                           float* __restrict__ scale, float* __restrict__ shift, float M) {
  const int i = threadIdx.x;
  if (i < 64) {
    const float mu = gsum[i] / M;
    float var = gssq[i] / M - mu * mu;
    var = fmaxf(var, 0.f);
    const float sc = g[i] * rsqrtf(var + EPS);
    scale[i] = sc;
    shift[i] = be[i] - mu * sc;
  }
}

// ---------- zero accumulators ----------
__global__ void k_zero(float* __restrict__ out, int nOut, int* __restrict__ cnt, int N,
                       float* __restrict__ stats) {
  const int i = blockIdx.x * blockDim.x + threadIdx.x;
  const int s = gridDim.x * blockDim.x;
  for (int j = i; j < nOut; j += s) out[j] = 0.f;
  for (int j = i; j < N; j += s) cnt[j] = 0;
  if (i < 512) stats[i] = 0.f;
}

extern "C" void kernel_launch(void* const* d_in, const int* in_sizes, int n_in,
                              void* d_out, int out_size, void* d_ws, size_t ws_size,
                              hipStream_t stream) {
  (void)n_in; (void)out_size;
  const float* x   = (const float*)d_in[0];
  const int*   ei  = (const int*)d_in[1];
  const float* ea  = (const float*)d_in[2];
  const float* W1a = (const float*)d_in[5];
  const float* b1a = (const float*)d_in[6];
  const float* g1  = (const float*)d_in[7];
  const float* be1 = (const float*)d_in[8];
  const float* W2a = (const float*)d_in[9];
  const float* b2a = (const float*)d_in[10];
  const float* W1b = (const float*)d_in[11];
  const float* b1b = (const float*)d_in[12];
  const float* g2  = (const float*)d_in[13];
  const float* be2 = (const float*)d_in[14];
  const float* W2b = (const float*)d_in[15];
  const float* b2b = (const float*)d_in[16];

  const int N = in_sizes[0] / 64;   // x: [N,64]
  const int E = in_sizes[2] / 64;   // edge_attr: [E,64]
  const int* rowp = ei;
  const int* colp = ei + E;
  const int Ef = E & ~15, Nf = N & ~15;

  float* ws    = (float*)d_ws;
  float* gsum1 = ws;        float* gssq1 = ws + 64;
  float* sc1   = ws + 128;  float* sh1   = ws + 192;
  float* gsum2 = ws + 256;  float* gssq2 = ws + 320;
  float* sc2   = ws + 384;  float* sh2   = ws + 448;
  int*   cnt   = (int*)(ws + 512);
  float* outp  = (float*)d_out;     // also the scatter-sum accumulator

  // optional f16 h1 cache in workspace (runtime-constant choice)
  const size_t h1off = (512u * 4 + (size_t)N * 4 + 255) & ~(size_t)255;
  const bool cached = ws_size >= h1off + (size_t)E * 64 * sizeof(_Float16);
  _Float16* h1buf = (_Float16*)((char*)d_ws + h1off);

  k_zero<<<2048, 256, 0, stream>>>(outp, N * 64, cnt, N, ws);
  // pass 1: edge layer-1 stats (+ h1 f16 spill when workspace allows)
  if (cached)
    k_edge<1><<<1024, TB, 0, stream>>>(x, rowp, colp, ea, W1a, b1a,
                                       nullptr, nullptr, nullptr, nullptr,
                                       gsum1, gssq1, nullptr, nullptr, h1buf, Ef);
  else
    k_edge<0><<<1024, TB, 0, stream>>>(x, rowp, colp, ea, W1a, b1a,
                                       nullptr, nullptr, nullptr, nullptr,
                                       gsum1, gssq1, nullptr, nullptr, nullptr, Ef);
  if (E != Ef)
    k_edge_tail<false><<<1, 64, 0, stream>>>(x, rowp, colp, ea, W1a, b1a,
                                             nullptr, nullptr, nullptr, nullptr,
                                             gsum1, gssq1, nullptr, nullptr, Ef, E);
  k_finalize<<<1, 64, 0, stream>>>(gsum1, gssq1, g1, be1, sc1, sh1, (float)E);
  // pass 2: BN+ReLU+layer2+scatter (cached h1 fast path, or full recompute)
  if (cached)
    k_edge2<<<1024, TB, 0, stream>>>(h1buf, colp, W2a, b2a, sc1, sh1,
                                     outp, cnt, Ef);
  else
    k_edge<2><<<1024, TB, 0, stream>>>(x, rowp, colp, ea, W1a, b1a, W2a, b2a,
                                       sc1, sh1, nullptr, nullptr, outp, cnt,
                                       nullptr, Ef);
  if (E != Ef)
    k_edge_tail<true><<<1, 64, 0, stream>>>(x, rowp, colp, ea, W1a, b1a, W2a, b2a,
                                            sc1, sh1, nullptr, nullptr, outp, cnt, Ef, E);
  // pass 3: node layer-1 stats
  k_node<false><<<512, TB, 0, stream>>>(x, outp, cnt, W1b, b1b,
                                        nullptr, nullptr, nullptr, nullptr,
                                        gsum2, gssq2, nullptr, Nf);
  if (N != Nf)
    k_node_tail<false><<<1, 64, 0, stream>>>(x, outp, cnt, W1b, b1b,
                                             nullptr, nullptr, nullptr, nullptr,
                                             gsum2, gssq2, nullptr, Nf, N);
  k_finalize<<<1, 64, 0, stream>>>(gsum2, gssq2, g2, be2, sc2, sh2, (float)N);
  // pass 4: full node MLP, in-place over the accumulator
  k_node<true><<<512, TB, 0, stream>>>(x, outp, cnt, W1b, b1b, W2b, b2b,
                                       sc2, sh2, nullptr, nullptr, outp, Nf);
  if (N != Nf)
    k_node_tail<true><<<1, 64, 0, stream>>>(x, outp, cnt, W1b, b1b, W2b, b2b,
                                            sc2, sh2, nullptr, nullptr, outp, Nf, N);
}
